// PoolingByMultiheadAttention_90701119357362
// MI455X (gfx1250) — matrix operand order repaired and checked by reading, not compiled, not run on hardware
//
#include <hip/hip_runtime.h>
#include <math.h>

#define D      256
#define H      8
#define DH     32
#define BGRAPH 4096
#define DFF    1024
#define EPSLN  1e-5f

typedef __attribute__((ext_vector_type(16))) _Float16 v16h;
typedef __attribute__((ext_vector_type(8)))  _Float16 v8h;
typedef __attribute__((ext_vector_type(8)))  float    v8f;

// ---------------------------------------------------------------------------
// K0: qv = Wq @ seed + bq ; P[h][:] = scale * sum_{i in head h} qv[i]*Wk[i][:]
// (per-head constant from bk cancels in the softmax, so it is dropped)
// ---------------------------------------------------------------------------
__global__ __launch_bounds__(256) void k_prep(const float* __restrict__ seeds,
                                              const float* __restrict__ Wq,
                                              const float* __restrict__ bq,
                                              const float* __restrict__ Wk,
                                              float* __restrict__ P) {
  __shared__ float sSh[D];
  __shared__ float qvSh[D];
  const int t = threadIdx.x;
  sSh[t] = seeds[t];
  __syncthreads();
  float acc = bq[t];
  for (int j = 0; j < D; ++j) acc += sSh[j] * Wq[t * D + j];
  qvSh[t] = acc;
  __syncthreads();
  const float scale = 0.17677669529663687f;  // 1/sqrt(DH)
  for (int h = 0; h < H; ++h) {
    float p = 0.f;
    for (int i = 0; i < DH; ++i) p += qvSh[h * DH + i] * Wk[(h * DH + i) * D + t];
    P[h * D + t] = p * scale;
  }
}

// ---------------------------------------------------------------------------
// K1: segment offsets from sorted batch[] -> off[0..B] (CSR style)
// ---------------------------------------------------------------------------
__global__ void k_offsets(const int* __restrict__ batch, int* __restrict__ off, int n) {
  const int i = blockIdx.x * blockDim.x + threadIdx.x;
  if (i >= n) return;
  const int bc = batch[i];
  if (i == 0) {
    for (int b = 0; b <= bc; ++b) off[b] = 0;
  } else {
    const int bp = batch[i - 1];
    for (int b = bp + 1; b <= bc; ++b) off[b] = i;
  }
  if (i == n - 1) {
    for (int b = bc + 1; b <= BGRAPH; ++b) off[b] = n;
  }
}

// ---------------------------------------------------------------------------
// K2: fp32 -> f16 conversion (weights)
// ---------------------------------------------------------------------------
__global__ void k_convert(const float* __restrict__ src, _Float16* __restrict__ dst, int n) {
  const int i = blockIdx.x * blockDim.x + threadIdx.x;
  if (i < n) dst[i] = (_Float16)src[i];
}

// ---------------------------------------------------------------------------
// helpers
// ---------------------------------------------------------------------------
__device__ inline float block_reduce_sum_256(float v, float* red) {
  const int t = threadIdx.x;
  red[t] = v;
  __syncthreads();
  for (int s = 128; s > 0; s >>= 1) {
    if (t < s) red[t] += red[t + s];
    __syncthreads();
  }
  const float r = red[0];
  __syncthreads();
  return r;
}

// ---------------------------------------------------------------------------
// K3: per-graph pooled attention. One block (8 wave32) per graph.
// Streaming online-softmax over the graph's contiguous node range; the V
// projection is hoisted out of the node loop via
//   ctx_h = (Wv_h @ sum_n e_nh * x_n + bv_h * den_h) / max(den_h,1e-20)
// Then Wo projection, residual with seed, LayerNorm1 -> y (fp32) and yh (f16).
// ---------------------------------------------------------------------------
__global__ void k_pool(const float* __restrict__ x, const int* __restrict__ off,
                       const float* __restrict__ P, const float* __restrict__ Wv,
                       const float* __restrict__ bv, const float* __restrict__ Wo,
                       const float* __restrict__ bo, const float* __restrict__ seeds,
                       const float* __restrict__ g1, const float* __restrict__ b1,
                       float* __restrict__ y, _Float16* __restrict__ yh) {
  __shared__ float accSh[H][D];   // weighted sum of x per head
  __shared__ float denSh[H];
  __shared__ float mAll[8][H];
  __shared__ float ctxSh[D];
  __shared__ float red[256];

  const int b = blockIdx.x;
  const int t = threadIdx.x;
  const int w = t >> 5;
  const int lane = t & 31;
  const int s0 = off[b], s1 = off[b + 1];

  for (int i = t; i < H * D; i += 256) (&accSh[0][0])[i] = 0.f;
  if (t < H) denSh[t] = 0.f;
  __syncthreads();

  // per-lane slice of P: dims d = lane*8 + c
  float Pr[H][8];
#pragma unroll
  for (int h = 0; h < H; ++h)
#pragma unroll
    for (int c = 0; c < 8; ++c) Pr[h][c] = P[h * D + lane * 8 + c];

  float m[H], den[H], acc[H][8];
#pragma unroll
  for (int h = 0; h < H; ++h) {
    m[h] = -INFINITY;
    den[h] = 0.f;
#pragma unroll
    for (int c = 0; c < 8; ++c) acc[h][c] = 0.f;
  }

  for (int n = s0 + w; n < s1; n += 8) {
    const float4* xr = (const float4*)(x + (size_t)n * D + lane * 8);
    const float4 xa = xr[0];
    const float4 xb = xr[1];
    const float xv[8] = {xa.x, xa.y, xa.z, xa.w, xb.x, xb.y, xb.z, xb.w};

#pragma unroll
    for (int h = 0; h < H; ++h) {
      float p = 0.f;
#pragma unroll
      for (int c = 0; c < 8; ++c) p += xv[c] * Pr[h][c];
#pragma unroll
      for (int o = 16; o > 0; o >>= 1) p += __shfl_xor(p, o, 32);
      const float logit = p;           // wave-uniform
      if (logit > m[h]) {              // online-softmax rescale
        const float f = __expf(m[h] - logit);  // exp(-inf)=0 on first hit
        den[h] *= f;
#pragma unroll
        for (int c = 0; c < 8; ++c) acc[h][c] *= f;
        m[h] = logit;
      }
      const float e = __expf(logit - m[h]);
      den[h] += e;
#pragma unroll
      for (int c = 0; c < 8; ++c) acc[h][c] += e * xv[c];
    }
  }

  // merge the 8 waves (rebase to global max, LDS float atomics)
#pragma unroll
  for (int h = 0; h < H; ++h) mAll[w][h] = m[h];
  __syncthreads();
#pragma unroll
  for (int h = 0; h < H; ++h) {
    float M = mAll[0][h];
#pragma unroll
    for (int w2 = 1; w2 < 8; ++w2) M = fmaxf(M, mAll[w2][h]);
    const float f = (m[h] == -INFINITY) ? 0.f : __expf(m[h] - M);
    if (lane == 0) atomicAdd(&denSh[h], den[h] * f);
#pragma unroll
    for (int c = 0; c < 8; ++c) atomicAdd(&accSh[h][lane * 8 + c], acc[h][c] * f);
  }
  __syncthreads();

  // ctx = (Wv_h @ Xbar + bv*den)/max(den,1e-20)   (thread t owns output dim t)
  const int hh = t >> 5;
  float num = 0.f;
  for (int j = 0; j < D; ++j) num += Wv[(size_t)t * D + j] * accSh[hh][j];
  const float dh = denSh[hh];
  ctxSh[t] = (num + bv[t] * dh) / fmaxf(dh, 1e-20f);
  __syncthreads();

  // attn = Wo @ ctx + bo ; residual with seed ; LayerNorm1
  float at = bo[t];
  for (int j = 0; j < D; ++j) at += Wo[(size_t)t * D + j] * ctxSh[j];
  const float r = seeds[t] + at;

  const float mu = block_reduce_sum_256(r, red) * (1.f / D);
  const float dv = r - mu;
  const float var = block_reduce_sum_256(dv * dv, red) * (1.f / D);
  const float yv = dv * rsqrtf(var + EPSLN) * g1[t] + b1[t];

  y[(size_t)b * D + t] = yv;
  yh[(size_t)b * D + t] = (_Float16)yv;
}

// ---------------------------------------------------------------------------
// WMMA fragment loads per CDNA5 ISA layouts (wave32, 16x16x32 f16):
//  A (MxK): lane<16 -> row M=lane, K {k..k+7, k+16..k+23}; lanes>=16 -> +8 K offset
//  B (KxN): lane<16 -> col N=lane, K {k..k+7, k+16..k+23}; lanes>=16 -> +8 K offset
//  D (MxN): VGPR r -> M = r + (lane>=16 ? 8 : 0), N = lane&15
// Both A and B fragments are two contiguous 16B loads from row-major f16.
// ---------------------------------------------------------------------------
__device__ inline v16h load_a_frag(const _Float16* p) {
  const v8h lo = *(const v8h*)p;
  const v8h hi = *(const v8h*)(p + 16);
  return __builtin_shufflevector(lo, hi, 0, 1, 2, 3, 4, 5, 6, 7, 8, 9, 10, 11, 12, 13, 14, 15);
}
__device__ inline v16h load_b_frag(const _Float16* p) {
  const v8h lo = *(const v8h*)p;
  const v8h hi = *(const v8h*)(p + 16);
  return __builtin_shufflevector(lo, hi, 0, 1, 2, 3, 4, 5, 6, 7, 8, 9, 10, 11, 12, 13, 14, 15);
}

// K4: H = relu(y @ W1^T + c1)  [4096 x 1024], f16 out. One wave per 16x16 tile.
__global__ __launch_bounds__(256) void k_gemm_ffn1(const _Float16* __restrict__ A,
                                                   const _Float16* __restrict__ Bm,
                                                   const float* __restrict__ bias,
                                                   _Float16* __restrict__ Out) {
  const int lane = threadIdx.x & 31;
  const int wv = threadIdx.x >> 5;
  const int tile = blockIdx.x * 8 + wv;
  const int TN = DFF / 16;
  const int tm = tile / TN, tn = tile % TN;
  const int lh = lane & 15, hsel = lane >> 4;

  const _Float16* Arow = A + (size_t)(tm * 16 + lh) * D + hsel * 8;
  const _Float16* Brow = Bm + (size_t)(tn * 16 + lh) * D + hsel * 8;

  v8f c = {0.f, 0.f, 0.f, 0.f, 0.f, 0.f, 0.f, 0.f};
#pragma unroll
  for (int k = 0; k < D; k += 32) {
    const v16h a = load_a_frag(Arow + k);
    const v16h bb = load_b_frag(Brow + k);
    c = __builtin_amdgcn_wmma_f32_16x16x32_f16(false, a, false, bb, (short)0, c, false, false);
  }
  const int col = tn * 16 + lh;
  const float bz = bias[col];
  const int row0 = tm * 16 + hsel * 8;
#pragma unroll
  for (int r = 0; r < 8; ++r) {
    const float v = fmaxf(c[r] + bz, 0.f);
    Out[(size_t)(row0 + r) * DFF + col] = (_Float16)v;
  }
}

// K5: out_pre = H @ W2^T + c2  [4096 x 256], fp32 out. One wave per 16x16 tile.
__global__ __launch_bounds__(256) void k_gemm_ffn2(const _Float16* __restrict__ A,
                                                   const _Float16* __restrict__ Bm,
                                                   const float* __restrict__ bias,
                                                   float* __restrict__ Out) {
  const int lane = threadIdx.x & 31;
  const int wv = threadIdx.x >> 5;
  const int tile = blockIdx.x * 8 + wv;
  const int TN = D / 16;
  const int tm = tile / TN, tn = tile % TN;
  const int lh = lane & 15, hsel = lane >> 4;

  const _Float16* Arow = A + (size_t)(tm * 16 + lh) * DFF + hsel * 8;
  const _Float16* Brow = Bm + (size_t)(tn * 16 + lh) * DFF + hsel * 8;

  v8f c = {0.f, 0.f, 0.f, 0.f, 0.f, 0.f, 0.f, 0.f};
#pragma unroll
  for (int k = 0; k < DFF; k += 32) {
    const v16h a = load_a_frag(Arow + k);
    const v16h bb = load_b_frag(Brow + k);
    c = __builtin_amdgcn_wmma_f32_16x16x32_f16(false, a, false, bb, (short)0, c, false, false);
  }
  const int col = tn * 16 + lh;
  const float bz = bias[col];
  const int row0 = tm * 16 + hsel * 8;
#pragma unroll
  for (int r = 0; r < 8; ++r) Out[(size_t)(row0 + r) * D + col] = c[r] + bz;
}

// K6: out = LayerNorm2(y + ffn)  -> d_out
__global__ __launch_bounds__(256) void k_final(const float* __restrict__ y,
                                               const float* __restrict__ fp,
                                               const float* __restrict__ g2,
                                               const float* __restrict__ b2,
                                               float* __restrict__ out) {
  __shared__ float red[256];
  const int b = blockIdx.x, t = threadIdx.x;
  const float r = y[(size_t)b * D + t] + fp[(size_t)b * D + t];
  const float mu = block_reduce_sum_256(r, red) * (1.f / D);
  const float dv = r - mu;
  const float var = block_reduce_sum_256(dv * dv, red) * (1.f / D);
  out[(size_t)b * D + t] = dv * rsqrtf(var + EPSLN) * g2[t] + b2[t];
}

// ---------------------------------------------------------------------------
extern "C" void kernel_launch(void* const* d_in, const int* in_sizes, int n_in,
                              void* d_out, int out_size, void* d_ws, size_t ws_size,
                              hipStream_t stream) {
  const float* x     = (const float*)d_in[0];
  const int*   batch = (const int*)d_in[1];
  const float* seeds = (const float*)d_in[3];
  const float* Wq    = (const float*)d_in[4];
  const float* bq    = (const float*)d_in[5];
  const float* Wk    = (const float*)d_in[6];
  /* bk (d_in[7]) cancels inside the softmax */
  const float* Wv    = (const float*)d_in[8];
  const float* bv    = (const float*)d_in[9];
  const float* Wo    = (const float*)d_in[10];
  const float* bo    = (const float*)d_in[11];
  const float* g1    = (const float*)d_in[12];
  const float* b1    = (const float*)d_in[13];
  const float* W1    = (const float*)d_in[14];
  const float* c1    = (const float*)d_in[15];
  const float* W2    = (const float*)d_in[16];
  const float* c2    = (const float*)d_in[17];
  const float* g2    = (const float*)d_in[18];
  const float* b2    = (const float*)d_in[19];
  const int N = in_sizes[0] / D;

  char* base = (char*)d_ws;
  size_t cur = 0;
  auto alloc = [&](size_t bytes) -> char* {
    char* p = base + cur;
    cur = (cur + bytes + 255) & ~(size_t)255;
    return p;
  };
  float*    P    = (float*)alloc((size_t)H * D * sizeof(float));
  int*      off  = (int*)alloc((size_t)(BGRAPH + 1) * sizeof(int));
  float*    y    = (float*)alloc((size_t)BGRAPH * D * sizeof(float));
  _Float16* yh   = (_Float16*)alloc((size_t)BGRAPH * D * sizeof(_Float16));
  _Float16* W1h  = (_Float16*)alloc((size_t)DFF * D * sizeof(_Float16));
  _Float16* W2h  = (_Float16*)alloc((size_t)D * DFF * sizeof(_Float16));
  _Float16* Hh   = (_Float16*)alloc((size_t)BGRAPH * DFF * sizeof(_Float16));
  float*    outp = (float*)alloc((size_t)BGRAPH * D * sizeof(float));

  hipLaunchKernelGGL(k_prep, dim3(1), dim3(256), 0, stream, seeds, Wq, bq, Wk, P);
  hipLaunchKernelGGL(k_offsets, dim3((N + 255) / 256), dim3(256), 0, stream, batch, off, N);
  hipLaunchKernelGGL(k_convert, dim3((DFF * D + 255) / 256), dim3(256), 0, stream, W1, W1h, DFF * D);
  hipLaunchKernelGGL(k_convert, dim3((D * DFF + 255) / 256), dim3(256), 0, stream, W2, W2h, D * DFF);
  hipLaunchKernelGGL(k_pool, dim3(BGRAPH), dim3(256), 0, stream,
                     x, off, P, Wv, bv, Wo, bo, seeds, g1, b1, y, yh);
  hipLaunchKernelGGL(k_gemm_ffn1, dim3((BGRAPH / 16) * (DFF / 16) / 8), dim3(256), 0, stream,
                     yh, W1h, c1, Hh);
  hipLaunchKernelGGL(k_gemm_ffn2, dim3((BGRAPH / 16) * (D / 16) / 8), dim3(256), 0, stream,
                     Hh, W2h, c2, outp);
  hipLaunchKernelGGL(k_final, dim3(BGRAPH), dim3(256), 0, stream, y, outp, g2, b2, (float*)d_out);
}
